// CrossAttentionBlock_2388001817002
// MI455X (gfx1250) — compile-verified, matrix operand order checked by reading, not run on hardware
//
#include <hip/hip_runtime.h>
#include <hip/hip_bf16.h>
#include <math.h>

// ---------------------------------------------------------------------------
// CDNA5 / gfx1250 cross-attention block. wave32, WMMA bf16 16x16x32,
// async global->LDS staging (ASYNCcnt) with double-buffered LDS.
// ---------------------------------------------------------------------------

typedef __bf16 bf16;
typedef bf16  v16bf __attribute__((ext_vector_type(16)));
typedef bf16  v8bf  __attribute__((ext_vector_type(8)));
typedef float v8f   __attribute__((ext_vector_type(8)));

#define WMMA_BF16(a, b, c)                                                  \
  __builtin_amdgcn_wmma_f32_16x16x32_bf16(false, (a), false, (b),           \
                                          (short)0, (c), false, false)

#if __has_builtin(__builtin_amdgcn_s_wait_asynccnt)
#define WAIT_ASYNC(n) do { __builtin_amdgcn_s_wait_asynccnt(n); \
                           asm volatile("" ::: "memory"); } while (0)
#else
#define WAIT_ASYNC(n) asm volatile("s_wait_asynccnt %0" :: "i"(n) : "memory")
#endif

// Async copy of 16 bytes/lane from global (saddr + voffset) straight to LDS.
// GVS addressing: mem_addr = SGPR_U64 + VGPR_I32 (+ inst offset 0).
// lds_off is the per-lane LDS byte address (low 32 bits of generic pointer).
__device__ __forceinline__ void async_ld16(unsigned lds_off,
                                           const void* gbase, int voff) {
  asm volatile("global_load_async_to_lds_b128 %0, %1, %2"
               :: "v"(lds_off), "v"(voff), "s"(gbase)
               : "memory");
}

// Load a 16x32 bf16 A-fragment (or 32x16 B-fragment, symmetric layout) from
// a row-major tile with row stride `rstride` (elements).
// Layout per ISA 7.12.2: lanes 0-15 -> row = lane, K elems {0..7,16..23};
// lanes 16-31 -> row = lane-16, K elems {8..15,24..31}.
__device__ __forceinline__ v16bf frag_ld(const bf16* p, int lane, int rstride) {
  const int row  = lane & 15;
  const int half = (lane >> 4) & 1;
  const bf16* q = p + (long)row * rstride + half * 8;
  v8bf lo = *(const v8bf*)(q);
  v8bf hi = *(const v8bf*)(q + 16);
  return __builtin_shufflevector(lo, hi, 0, 1, 2, 3, 4, 5, 6, 7,
                                 8, 9, 10, 11, 12, 13, 14, 15);
}

__device__ __forceinline__ v8f zero8() {
  v8f z;
#pragma unroll
  for (int e = 0; e < 8; ++e) z[e] = 0.0f;
  return z;
}

// ---------------------------------------------------------------------------
// Weight convert + transpose:  out[n*K + k] = (bf16) in[k*N + n]
// ---------------------------------------------------------------------------
__global__ void __launch_bounds__(256)
wconv_t(const float* __restrict__ in, bf16* __restrict__ out, int K, int Ncols) {
  long i = (long)blockIdx.x * 256 + threadIdx.x;
  long total = (long)K * Ncols;
  if (i >= total) return;
  long n = i / K, k = i - n * K;
  out[i] = (bf16)in[k * (long)Ncols + n];
}

// ---------------------------------------------------------------------------
// Row LayerNorm (C = 1024), f32 in -> bf16 out, one block per row.
// ---------------------------------------------------------------------------
__global__ void __launch_bounds__(256)
ln_rows(const float* __restrict__ x, const float* __restrict__ w,
        const float* __restrict__ bp, bf16* __restrict__ out, int C) {
  long row = blockIdx.x;
  const float* xr = x + row * (long)C;
  float s = 0.f, s2 = 0.f;
  for (int i = threadIdx.x; i < C; i += 256) {
    float v = xr[i];
    s += v; s2 += v * v;
  }
#pragma unroll
  for (int o = 16; o > 0; o >>= 1) {
    s  += __shfl_xor(s,  o, 32);
    s2 += __shfl_xor(s2, o, 32);
  }
  __shared__ float ps[8], ps2[8];
  __shared__ float mu_s, rstd_s;
  int wv = threadIdx.x >> 5;
  if ((threadIdx.x & 31) == 0) { ps[wv] = s; ps2[wv] = s2; }
  __syncthreads();
  if (threadIdx.x == 0) {
    float t = 0.f, t2 = 0.f;
#pragma unroll
    for (int i = 0; i < 8; ++i) { t += ps[i]; t2 += ps2[i]; }
    float mu = t / C;
    float var = t2 / C - mu * mu;
    mu_s = mu; rstd_s = rsqrtf(var + 1e-5f);
  }
  __syncthreads();
  float mu = mu_s, rstd = rstd_s;
  bf16* orow = out + row * (long)C;
  for (int i = threadIdx.x; i < C; i += 256)
    orow[i] = (bf16)((xr[i] - mu) * rstd * w[i] + bp[i]);
}

// ---------------------------------------------------------------------------
// Generic WMMA bf16 GEMM:  out = A(rows x K, bf16) * Wt(Ncols x K, bf16)^T
// 256 threads / 8 waves; block tile 128x128; wave tile 64x32; K step 32.
// Double-buffered LDS fed by GLOBAL_LOAD_ASYNC_TO_LDS_B128 (ASYNCcnt),
// so the next tile streams in while the current one is in the WMMAs.
// Epilogue: optional bias, exact GELU, residual; f32 and/or bf16 stores.
// ---------------------------------------------------------------------------
#define LSTR 40  // padded LDS row stride (bf16 elems) -> conflict-free b128 reads

__global__ void __launch_bounds__(256)
gemm_wmma_bf16(const bf16* __restrict__ A, const bf16* __restrict__ Wt,
               const float* __restrict__ bias, const float* __restrict__ resid,
               float* __restrict__ outF, bf16* __restrict__ outB,
               int K, int Ncols, int gelu_flag) {
  __shared__ bf16 As[2][128 * LSTR];
  __shared__ bf16 Bs[2][128 * LSTR];

  const int tid  = threadIdx.x;
  const int lane = tid & 31;
  const int wave = tid >> 5;
  const int wm   = wave & 1;   // 2 wave-rows
  const int wn   = wave >> 1;  // 4 wave-cols
  const long rowBase = (long)blockIdx.y * 128;
  const long colBase = (long)blockIdx.x * 128;

  v8f acc[4][2];
#pragma unroll
  for (int i = 0; i < 4; ++i)
#pragma unroll
    for (int j = 0; j < 2; ++j) acc[i][j] = zero8();

  const int lr  = tid >> 1;         // tile row handled by this thread (0..127)
  const int seg = (tid & 1) * 16;   // k-segment (0 or 16 elements)

  // Per-lane byte offsets into A / Wt (fits 32-bit: max 32 MB).
  const int aoff = (int)(((rowBase + lr) * (long)K + seg) * 2);
  const int boff = (int)(((colBase + lr) * (long)K + seg) * 2);
  // Per-lane LDS byte addresses (low 32 bits of generic pointer = LDS offset).
  unsigned lda[2], ldb[2];
  lda[0] = (unsigned)(size_t)&As[0][lr * LSTR + seg];
  lda[1] = (unsigned)(size_t)&As[1][lr * LSTR + seg];
  ldb[0] = (unsigned)(size_t)&Bs[0][lr * LSTR + seg];
  ldb[1] = (unsigned)(size_t)&Bs[1][lr * LSTR + seg];

  // Stage one 128x32 A-tile + 128x32 B-tile into buffer `buf`:
  // 4 async instructions per wave (2 x 16B per thread per tile).
  auto stage = [&](int buf, int kb) {
    const int ka = aoff + kb * 2;
    const int kw = boff + kb * 2;
    async_ld16(lda[buf],      A,  ka);
    async_ld16(lda[buf] + 16, A,  ka + 16);
    async_ld16(ldb[buf],      Wt, kw);
    async_ld16(ldb[buf] + 16, Wt, kw + 16);
  };

  const int nsteps = K >> 5;
  stage(0, 0);  // prologue prefetch

  for (int s = 0; s < nsteps; ++s) {
    const int p = s & 1;
    if (s + 1 < nsteps) {
      stage(p ^ 1, (s + 1) * 32);  // overlap: next tile in flight
      WAIT_ASYNC(4);               // current buffer's 4 asyncs done
    } else {
      WAIT_ASYNC(0);
    }
    __syncthreads();

    v16bf af[4], bfr[2];
#pragma unroll
    for (int fr = 0; fr < 4; ++fr)
      af[fr] = frag_ld(&As[p][(wm * 64 + fr * 16) * LSTR], lane, LSTR);
#pragma unroll
    for (int fc = 0; fc < 2; ++fc)
      bfr[fc] = frag_ld(&Bs[p][(wn * 32 + fc * 16) * LSTR], lane, LSTR);

#pragma unroll
    for (int fr = 0; fr < 4; ++fr)
#pragma unroll
      for (int fc = 0; fc < 2; ++fc)
        acc[fr][fc] = WMMA_BF16(af[fr], bfr[fc], acc[fr][fc]);
    __syncthreads();  // everyone done reading buf p before it is re-staged
  }

  // Epilogue. C layout: elem r -> row (half*8 + r), col (lane & 15).
  const int ln_  = lane & 15;
  const int half = (lane >> 4) & 1;
#pragma unroll
  for (int fr = 0; fr < 4; ++fr) {
#pragma unroll
    for (int fc = 0; fc < 2; ++fc) {
      long col = colBase + wn * 32 + fc * 16 + ln_;
      float bv = bias ? bias[col] : 0.0f;
#pragma unroll
      for (int r = 0; r < 8; ++r) {
        long row = rowBase + wm * 64 + fr * 16 + half * 8 + r;
        float v = acc[fr][fc][r] + bv;
        if (gelu_flag) v = 0.5f * v * (1.0f + erff(v * 0.70710678118f));
        long idx = row * (long)Ncols + col;
        if (resid) v += resid[idx];
        if (outF) outF[idx] = v;
        if (outB) outB[idx] = (bf16)v;
      }
    }
  }
}

// ---------------------------------------------------------------------------
// Flash cross-attention. One wave per 16 query rows of one (b,h).
// q: (B,N,1024) bf16, col h*64+d.  kv: (B,M,2048) bf16, k at h*64+d,
// v at 1024+h*64+d.  mask: (B,N,M) u8 (nonzero => -inf).  out: (B,N,1024) bf16.
// ---------------------------------------------------------------------------
__global__ void __launch_bounds__(128)
attn_flash(const bf16* __restrict__ qb, const bf16* __restrict__ kvb,
           const unsigned char* __restrict__ mask, bf16* __restrict__ outb) {
  const int Nn = 2048, Mm = 2048, Cc = 1024;
  __shared__ bf16 Pl[4][16 * LSTR];

  const int lane = threadIdx.x & 31;
  const int wave = threadIdx.x >> 5;
  const int id = blockIdx.x * 4 + wave;     // 0..4095
  const int nt = id & 127;                  // N/16 tiles
  const int h  = (id >> 7) & 15;
  const int b  = id >> 11;
  const int n0 = nt * 16;
  const float scale = 0.125f;               // D^-0.5, D=64
  const int ln_  = lane & 15;
  const int half = (lane >> 4) & 1;

  // Q fragments (16 x 64 = two 16x32 A-frags), loaded once.
  const bf16* qp = qb + ((long)(b * Nn + n0)) * Cc + h * 64;
  v16bf qa0 = frag_ld(qp,      lane, Cc);
  v16bf qa1 = frag_ld(qp + 32, lane, Cc);

  float mrun[8], lrun[8];
  v8f o[4];
#pragma unroll
  for (int r = 0; r < 8; ++r) { mrun[r] = -3.0e38f; lrun[r] = 0.0f; }
#pragma unroll
  for (int dg = 0; dg < 4; ++dg) o[dg] = zero8();

  const bf16* kbase = kvb + ((long)b * Mm) * 2048 + h * 64;
  const bf16* vbase = kvb + ((long)b * Mm) * 2048 + 1024 + h * 64;
  const unsigned char* mrow = mask + ((long)b * Nn + n0) * (long)Mm;

  for (int m0 = 0; m0 < Mm; m0 += 32) {
    // S = Q K^T for a 16x32 chunk of key columns (two 16x16 C-frags).
    v8f S[2];
#pragma unroll
    for (int g = 0; g < 2; ++g) {
      const bf16* kp = kbase + (long)(m0 + g * 16) * 2048;
      v16bf kb0 = frag_ld(kp,      lane, 2048);
      v16bf kb1 = frag_ld(kp + 32, lane, 2048);
      v8f c = zero8();
      c = WMMA_BF16(qa0, kb0, c);
      c = WMMA_BF16(qa1, kb1, c);
      S[g] = c;
    }

    // Scale, mask, online softmax (row-wise over this lane's 16-lane group).
#pragma unroll
    for (int r = 0; r < 8; ++r) {
      const unsigned char* mr = mrow + (long)(half * 8 + r) * Mm + m0;
      float v0 = S[0][r] * scale;
      float v1 = S[1][r] * scale;
      if (mr[ln_])      v0 = -3.0e38f;
      if (mr[16 + ln_]) v1 = -3.0e38f;

      float mx = fmaxf(v0, v1);
#pragma unroll
      for (int off = 8; off > 0; off >>= 1) mx = fmaxf(mx, __shfl_xor(mx, off, 16));
      float mn = fmaxf(mrun[r], mx);
      float alpha = __expf(mrun[r] - mn);
      float p0 = __expf(v0 - mn);
      float p1 = __expf(v1 - mn);
      float rs = p0 + p1;
#pragma unroll
      for (int off = 8; off > 0; off >>= 1) rs += __shfl_xor(rs, off, 16);
      lrun[r] = lrun[r] * alpha + rs;
      mrun[r] = mn;
#pragma unroll
      for (int dg = 0; dg < 4; ++dg) o[dg][r] *= alpha;

      // Stash P (C layout -> row-major LDS tile) for A-frag reload.
      Pl[wave][(half * 8 + r) * LSTR + ln_]      = (bf16)p0;
      Pl[wave][(half * 8 + r) * LSTR + 16 + ln_] = (bf16)p1;
    }
    asm volatile("" ::: "memory");  // keep LDS stores before reload; DS in-order/wave

    v16bf pa = frag_ld(&Pl[wave][0], lane, LSTR);

    // O += P (16x32) * V (32x64): four 16x16 output frags over D.
#pragma unroll
    for (int dg = 0; dg < 4; ++dg) {
      const bf16* vp = vbase + dg * 16 + ln_;
      v16bf vb;
#pragma unroll
      for (int e = 0; e < 16; ++e) {
        int k = (e & 7) + ((e >> 3) << 4) + half * 8;  // B-frag K index
        vb[e] = vp[(long)(m0 + k) * 2048];
      }
      o[dg] = WMMA_BF16(pa, vb, o[dg]);
    }
  }

  // Normalize and store bf16.
  bf16* op = outb + ((long)(b * Nn + n0)) * Cc + h * 64;
#pragma unroll
  for (int dg = 0; dg < 4; ++dg)
#pragma unroll
    for (int r = 0; r < 8; ++r) {
      float v = o[dg][r] / lrun[r];
      op[(long)(half * 8 + r) * Cc + dg * 16 + ln_] = (bf16)v;
    }
}

// ---------------------------------------------------------------------------
// Host launch
// ---------------------------------------------------------------------------
extern "C" void kernel_launch(void* const* d_in, const int* in_sizes, int n_in,
                              void* d_out, int out_size, void* d_ws, size_t ws_size,
                              hipStream_t stream) {
  (void)in_sizes; (void)n_in; (void)out_size; (void)ws_size;
  const int Bb = 2, Nn = 2048, Mm = 2048, Cc = 1024, HID = 4096;
  const long RQ = (long)Bb * Nn;   // 4096 query rows
  const long RK = (long)Bb * Mm;   // 4096 context rows

  const float* x      = (const float*)d_in[0];
  const float* ctx    = (const float*)d_in[1];
  const unsigned char* mask = (const unsigned char*)d_in[2];
  const float* qn_w   = (const float*)d_in[3];
  const float* qn_b   = (const float*)d_in[4];
  const float* cn_w   = (const float*)d_in[5];
  const float* cn_b   = (const float*)d_in[6];
  const float* n2_w   = (const float*)d_in[7];
  const float* n2_b   = (const float*)d_in[8];
  const float* q_w    = (const float*)d_in[9];
  const float* kv_w   = (const float*)d_in[10];
  const float* proj_w = (const float*)d_in[11];
  const float* proj_b = (const float*)d_in[12];
  const float* fc1_w  = (const float*)d_in[13];
  const float* fc1_b  = (const float*)d_in[14];
  const float* fc2_w  = (const float*)d_in[15];
  const float* fc2_b  = (const float*)d_in[16];
  float* out = (float*)d_out;

  char* ws = (char*)d_ws;
  size_t off = 0;
  auto wsalloc = [&](size_t bytes) -> void* {
    void* p = ws + off;
    off += (bytes + 255) & ~(size_t)255;
    return p;
  };
  bf16* wt_q    = (bf16*)wsalloc((size_t)Cc * Cc * 2);        // 2 MB  (C x C)^T
  bf16* wt_kv   = (bf16*)wsalloc((size_t)Cc * 2 * Cc * 2);    // 4 MB  (2C x C)
  bf16* wt_proj = (bf16*)wsalloc((size_t)Cc * Cc * 2);        // 2 MB
  bf16* wt_fc1  = (bf16*)wsalloc((size_t)Cc * HID * 2);       // 8 MB  (HID x C)
  bf16* wt_fc2  = (bf16*)wsalloc((size_t)HID * Cc * 2);       // 8 MB  (C x HID)
  bf16* xq      = (bf16*)wsalloc((size_t)RQ * Cc * 2);        // 8 MB
  bf16* ctxb    = (bf16*)wsalloc((size_t)RK * Cc * 2);        // 8 MB
  bf16* qbuf    = (bf16*)wsalloc((size_t)RQ * Cc * 2);        // 8 MB
  bf16* kvbuf   = (bf16*)wsalloc((size_t)RK * 2 * Cc * 2);    // 16 MB
  bf16* attno   = (bf16*)wsalloc((size_t)RQ * Cc * 2);        // 8 MB
  float* x1     = (float*)wsalloc((size_t)RQ * Cc * 4);       // 16 MB
  bf16* hln     = (bf16*)wsalloc((size_t)RQ * Cc * 2);        // 8 MB
  bf16* hact    = (bf16*)wsalloc((size_t)RQ * HID * 2);       // 32 MB

  // 1) weight convert + transpose (f32 KxN -> bf16 NxK)
  auto tlaunch = [&](const float* in, bf16* o, long K, long N) {
    long t = K * N;
    wconv_t<<<dim3((unsigned)((t + 255) / 256)), dim3(256), 0, stream>>>(in, o, (int)K, (int)N);
  };
  tlaunch(q_w,    wt_q,    Cc, Cc);
  tlaunch(kv_w,   wt_kv,   Cc, 2 * Cc);
  tlaunch(proj_w, wt_proj, Cc, Cc);
  tlaunch(fc1_w,  wt_fc1,  Cc, HID);
  tlaunch(fc2_w,  wt_fc2,  HID, Cc);

  // 2) pre-attention layernorms -> bf16
  ln_rows<<<dim3((unsigned)RQ), dim3(256), 0, stream>>>(x,   qn_w, qn_b, xq,   Cc);
  ln_rows<<<dim3((unsigned)RK), dim3(256), 0, stream>>>(ctx, cn_w, cn_b, ctxb, Cc);

  // 3) q = xq @ q_w            (4096 x 1024 x 1024), bf16 out
  gemm_wmma_bf16<<<dim3(Cc / 128, (unsigned)(RQ / 128)), dim3(256), 0, stream>>>(
      xq, wt_q, nullptr, nullptr, nullptr, qbuf, Cc, Cc, 0);
  // 4) kv = ctx @ kv_w         (4096 x 1024 x 2048), bf16 out (k|v packed)
  gemm_wmma_bf16<<<dim3(2 * Cc / 128, (unsigned)(RK / 128)), dim3(256), 0, stream>>>(
      ctxb, wt_kv, nullptr, nullptr, nullptr, kvbuf, Cc, 2 * Cc, 0);

  // 5) flash attention: B*H*(N/16) = 4096 wave-tiles, 4 waves/block
  attn_flash<<<dim3(1024), dim3(128), 0, stream>>>(qbuf, kvbuf, mask, attno);

  // 6) x1 = x + attno @ proj_w + proj_b   (f32 out)
  gemm_wmma_bf16<<<dim3(Cc / 128, (unsigned)(RQ / 128)), dim3(256), 0, stream>>>(
      attno, wt_proj, proj_b, x, x1, nullptr, Cc, Cc, 0);

  // 7) hln = LN(x1)
  ln_rows<<<dim3((unsigned)RQ), dim3(256), 0, stream>>>(x1, n2_w, n2_b, hln, Cc);

  // 8) hact = gelu(hln @ fc1_w + fc1_b)   (4096 x 1024 x 4096), bf16 out
  gemm_wmma_bf16<<<dim3(HID / 128, (unsigned)(RQ / 128)), dim3(256), 0, stream>>>(
      hln, wt_fc1, fc1_b, nullptr, nullptr, hact, Cc, HID, 1);

  // 9) out = x1 + hact @ fc2_w + fc2_b    (4096 x 4096 x 1024), f32 out
  gemm_wmma_bf16<<<dim3(Cc / 128, (unsigned)(RQ / 128)), dim3(256), 0, stream>>>(
      hact, wt_fc2, fc2_b, x1, out, nullptr, HID, Cc, 0);
}
